// SAGE_73778948211059
// MI455X (gfx1250) — matrix operand order repaired and checked by reading, not compiled, not run on hardware
//
#include <hip/hip_runtime.h>
#include <math.h>

#define KDIM 128

typedef __attribute__((ext_vector_type(16))) _Float16 v16h;
typedef __attribute__((ext_vector_type(8)))  float    v8f;

union Frag { v16h v; unsigned int u[8]; };

__device__ __forceinline__ unsigned int pack2h(float a, float b) {
    _Float16 ha = (_Float16)a, hb = (_Float16)b;
    unsigned short ua = __builtin_bit_cast(unsigned short, ha);
    unsigned short ub = __builtin_bit_cast(unsigned short, hb);
    return (unsigned int)ua | ((unsigned int)ub << 16);
}

// ================================================================ CSR build
__global__ void degree_int_kernel(const int* __restrict__ dst, int* __restrict__ degi, int E) {
    int e = blockIdx.x * blockDim.x + threadIdx.x;
    if (e < E) atomicAdd(&degi[dst[e]], 1);
}

__global__ void rdeg_kernel(const int* __restrict__ degi, float* __restrict__ rdeg, int n) {
    int i = blockIdx.x * blockDim.x + threadIdx.x;
    if (i < n) rdeg[i] = 1.0f / fmaxf((float)degi[i], 1.0f);
}

// exclusive scan, stage 1: per-256-block scan + block sums
__global__ void scan1_kernel(const int* __restrict__ degi, int* __restrict__ offs,
                             int* __restrict__ bsums, int n) {
    __shared__ int tmp[256];
    const int tid = threadIdx.x;
    const int i = blockIdx.x * 256 + tid;
    const int v = (i < n) ? degi[i] : 0;
    tmp[tid] = v;
    __syncthreads();
#pragma unroll
    for (int d = 1; d < 256; d <<= 1) {
        int t = (tid >= d) ? tmp[tid - d] : 0;
        __syncthreads();
        tmp[tid] += t;
        __syncthreads();
    }
    if (i < n) offs[i] = tmp[tid] - v;          // exclusive
    if (tid == 255) bsums[blockIdx.x] = tmp[255];
}

// stage 2: scan the block sums (B <= 256 for N <= 65536; serial fallback otherwise)
__global__ void scan2_kernel(int* __restrict__ bsums, int B) {
    __shared__ int tmp[256];
    const int tid = threadIdx.x;
    if (B <= 256) {
        const int v = (tid < B) ? bsums[tid] : 0;
        tmp[tid] = v;
        __syncthreads();
#pragma unroll
        for (int d = 1; d < 256; d <<= 1) {
            int t = (tid >= d) ? tmp[tid - d] : 0;
            __syncthreads();
            tmp[tid] += t;
            __syncthreads();
        }
        if (tid < B) bsums[tid] = tmp[tid] - v;  // exclusive
    } else if (tid == 0) {
        int run = 0;
        for (int i = 0; i < B; ++i) { int t = bsums[i]; bsums[i] = run; run += t; }
    }
}

// stage 3: add block offsets; also initialize bin cursors
__global__ void scan3_kernel(int* __restrict__ offs, const int* __restrict__ bsums,
                             int* __restrict__ cursor, int n) {
    int i = blockIdx.x * 256 + threadIdx.x;
    if (i < n) {
        int o = offs[i] + bsums[blockIdx.x];
        offs[i] = o;
        cursor[i] = o;
    }
}

__global__ void fill_adj_kernel(const int* __restrict__ src, const int* __restrict__ dst,
                                int* __restrict__ cursor, int* __restrict__ adj, int E) {
    int e = blockIdx.x * blockDim.x + threadIdx.x;
    if (e < E) {
        int p = atomicAdd(&cursor[dst[e]], 1);
        adj[p] = src[e];
    }
}

// ---------------------------------------------------- weight f32 -> f16 (row-padded)
__global__ void convert_w_kernel(const float* __restrict__ W, _Float16* __restrict__ Wh,
                                 int rows, int padRows) {
    int i = blockIdx.x * blockDim.x + threadIdx.x;
    int total = padRows * KDIM;
    if (i < total) {
        int r = i / KDIM;
        Wh[i] = (r < rows) ? (_Float16)W[i] : (_Float16)0.0f;
    }
}

// ======================================================= CSR mean aggregation
// One wave per destination node; 32 lanes x float4 = one 512B feature row.
// Pure gather (no atomics, no memset); mean scaling folded into the single store.
__global__ __launch_bounds__(256) void aggregate_csr_kernel(
    const float* __restrict__ h, const int* __restrict__ adj,
    const int* __restrict__ offs, const float* __restrict__ rdeg,
    float* __restrict__ agg, int n, int E)
{
    const int node = blockIdx.x * (blockDim.x >> 5) + (threadIdx.x >> 5);
    if (node >= n) return;
    const int c = (threadIdx.x & 31) * 4;
    const int beg = offs[node];
    const int end = (node + 1 < n) ? offs[node + 1] : E;

    float ax = 0.f, ay = 0.f, az = 0.f, aw = 0.f;
    int p = beg;
    for (; p + 1 < end; p += 2) {
        const int s0 = adj[p], s1 = adj[p + 1];
        const float4 v0 = *(const float4*)(h + (long)s0 * KDIM + c);
        const float4 v1 = *(const float4*)(h + (long)s1 * KDIM + c);
        ax += v0.x; ay += v0.y; az += v0.z; aw += v0.w;
        ax += v1.x; ay += v1.y; az += v1.z; aw += v1.w;
    }
    if (p < end) {
        const int s0 = adj[p];
        const float4 v0 = *(const float4*)(h + (long)s0 * KDIM + c);
        ax += v0.x; ay += v0.y; az += v0.z; aw += v0.w;
    }
    const float r = rdeg[node];
    float4 outv; outv.x = ax * r; outv.y = ay * r; outv.z = az * r; outv.w = aw * r;
    *(float4*)(agg + (long)node * KDIM + c) = outv;
}

// ------------------------------------------------- fused dual-GEMM + bias + LN + ReLU
// out[r, n] = act( LN( agg[r,:] @ Wl.T + bias + x[r,:] @ Wr.T ) )
// One wave per 16-row strip; 4 K-chunks of 32; OTILES column tiles of 16.
template <int OTILES, bool DOLN>
__global__ __launch_bounds__(256) void gemm_ln_kernel(
    const float* __restrict__ agg, const float* __restrict__ xin,
    const _Float16* __restrict__ Wl, const _Float16* __restrict__ Wr,
    const float* __restrict__ bias, const float* __restrict__ gamma,
    const float* __restrict__ beta,
    float* __restrict__ out, int n_nodes, int ocols, int ostride)
{
    const int lane = threadIdx.x & 31;
    const int wave = threadIdx.x >> 5;
    const int strip = blockIdx.x * 8 + wave;
    const int m_base = strip * 16;
    if (m_base >= n_nodes) return;

    const int ml = lane & 15;
    const int hi = lane >> 4;
    int row = m_base + ml;
    if (row >= n_nodes) row = n_nodes - 1;     // clamp: stores are guarded below

    // A fragments: 16x32 f16, lane = M (l%16), K interleaved per half-wave.
    Frag aA[4], aX[4];
#pragma unroll
    for (int kc = 0; kc < 4; ++kc) {
#pragma unroll
        for (int v = 0; v < 8; ++v) {
            const int k = kc * 32 + ((v & 4) ? 16 : 0) + hi * 8 + 2 * (v & 3);
            const float2 fa = *(const float2*)(agg + (long)row * KDIM + k);
            const float2 fx = *(const float2*)(xin + (long)row * KDIM + k);
            aA[kc].u[v] = pack2h(fa.x, fa.y);
            aX[kc].u[v] = pack2h(fx.x, fx.y);
        }
    }

    v8f   accs[OTILES];
    float mu[8], rstd[8];

#pragma unroll
    for (int t = 0; t < OTILES; ++t) {
        const int n = t * 16 + ml;
        v8f acc = {};
#pragma unroll
        for (int kc = 0; kc < 4; ++kc) {
            // B fragments: 32x16 f16, lane = N column, K interleaved per half-wave (matches A).
            Frag bL, bR;
#pragma unroll
            for (int v = 0; v < 8; ++v) {
                const int k = kc * 32 + ((v & 4) ? 16 : 0) + hi * 8 + 2 * (v & 3);
                bL.u[v] = *(const unsigned int*)(Wl + (long)n * KDIM + k);
                bR.u[v] = *(const unsigned int*)(Wr + (long)n * KDIM + k);
            }
            acc = __builtin_amdgcn_wmma_f32_16x16x32_f16(false, aA[kc].v, false, bL.v,
                                                         (short)0, acc, false, false);
            acc = __builtin_amdgcn_wmma_f32_16x16x32_f16(false, aX[kc].v, false, bR.v,
                                                         (short)0, acc, false, false);
        }
        const float bv = (n < ocols) ? bias[n] : 0.0f;
#pragma unroll
        for (int v = 0; v < 8; ++v) acc[v] += bv;
        accs[t] = acc;
    }

    if (DOLN) {
        // Row r = m_base + v + 8*hi lives in component v across all tiles,
        // spread over the 16 lanes of this half-wave.
        const float inv = 1.0f / (float)(OTILES * 16);
#pragma unroll
        for (int v = 0; v < 8; ++v) {
            float s = 0.0f, s2 = 0.0f;
#pragma unroll
            for (int t = 0; t < OTILES; ++t) { float x = accs[t][v]; s += x; s2 += x * x; }
#pragma unroll
            for (int m = 8; m >= 1; m >>= 1) { s += __shfl_xor(s, m); s2 += __shfl_xor(s2, m); }
            const float mean = s * inv;
            mu[v]   = mean;
            rstd[v] = rsqrtf(s2 * inv - mean * mean + 1e-5f);
        }
    }

#pragma unroll
    for (int t = 0; t < OTILES; ++t) {
        const int n = t * 16 + ml;
        float gv = 1.0f, bb = 0.0f;
        if (DOLN) { gv = gamma[n]; bb = beta[n]; }
#pragma unroll
        for (int v = 0; v < 8; ++v) {
            const int r = m_base + v + 8 * hi;   // C/D layout: lanes16-31 -> M = v+8
            float val = accs[t][v];
            if (DOLN) {
                val = (val - mu[v]) * rstd[v] * gv + bb;
                val = fmaxf(val, 0.0f);
            }
            if (r < n_nodes && n < ocols) out[(long)r * ostride + n] = val;
        }
    }
}

// ---------------------------------------------------------------- log_softmax (40)
__global__ void logsoftmax_kernel(float* __restrict__ out, int n, int ocols) {
    const int row = blockIdx.x * (blockDim.x >> 5) + (threadIdx.x >> 5);
    const int l = threadIdx.x & 31;
    if (row >= n) return;
    float* p = out + (long)row * ocols;
    const float v0 = p[l];                                   // l < 32 <= ocols
    const bool h1 = (l + 32) < ocols;
    const float v1 = h1 ? p[l + 32] : -3.402823e38f;
    float m = fmaxf(v0, v1);
#pragma unroll
    for (int k = 16; k >= 1; k >>= 1) m = fmaxf(m, __shfl_xor(m, k));
    float s = expf(v0 - m) + (h1 ? expf(v1 - m) : 0.0f);
#pragma unroll
    for (int k = 16; k >= 1; k >>= 1) s += __shfl_xor(s, k);
    const float lse = m + logf(s);
    p[l] = v0 - lse;
    if (h1) p[l + 32] = v1 - lse;
}

// =======================================================================
extern "C" void kernel_launch(void* const* d_in, const int* in_sizes, int n_in,
                              void* d_out, int out_size, void* d_ws, size_t ws_size,
                              hipStream_t stream) {
    const int N = in_sizes[0] / KDIM;     // 50000
    const int E = in_sizes[1] / 2;        // 800000
    const int OUT = in_sizes[9];          // bl3 length = 40
    const int OPAD = 48;                  // 3 column tiles of 16

    const float* x   = (const float*)d_in[0];
    const int*   ei  = (const int*)d_in[1];
    const int*   srcp = ei;
    const int*   dstp = ei + E;
    const float* Wl1 = (const float*)d_in[2];
    const float* bl1 = (const float*)d_in[3];
    const float* Wr1 = (const float*)d_in[4];
    const float* Wl2 = (const float*)d_in[5];
    const float* bl2 = (const float*)d_in[6];
    const float* Wr2 = (const float*)d_in[7];
    const float* Wl3 = (const float*)d_in[8];
    const float* bl3 = (const float*)d_in[9];
    const float* Wr3 = (const float*)d_in[10];
    const float* g1  = (const float*)d_in[11];
    const float* b1  = (const float*)d_in[12];
    const float* g2  = (const float*)d_in[13];
    const float* b2  = (const float*)d_in[14];

    const int B = (N + 255) / 256;        // scan blocks

    // workspace layout
    char* ws = (char*)d_ws;
    size_t o = 0;
    const size_t featB = (size_t)N * KDIM * sizeof(float);
    float* agg  = (float*)(ws + o); o += featB;
    float* h1b  = (float*)(ws + o); o += featB;
    float* h2b  = (float*)(ws + o); o += featB;
    float* rdeg = (float*)(ws + o); o += (size_t)N * sizeof(float);
    int* degi   = (int*)(ws + o);   o += (size_t)N * sizeof(int);
    int* offs   = (int*)(ws + o);   o += (size_t)(N + 1) * sizeof(int);
    int* cursor = (int*)(ws + o);   o += (size_t)N * sizeof(int);
    int* bsums  = (int*)(ws + o);   o += (size_t)B * sizeof(int);
    int* adj    = (int*)(ws + o);   o += (size_t)E * sizeof(int);
    o = (o + 255) & ~(size_t)255;
    const size_t wB  = (size_t)KDIM * KDIM * sizeof(_Float16);
    const size_t w3B = (size_t)OPAD * KDIM * sizeof(_Float16);
    _Float16* Wl1h = (_Float16*)(ws + o); o += wB;
    _Float16* Wr1h = (_Float16*)(ws + o); o += wB;
    _Float16* Wl2h = (_Float16*)(ws + o); o += wB;
    _Float16* Wr2h = (_Float16*)(ws + o); o += wB;
    _Float16* Wl3h = (_Float16*)(ws + o); o += w3B;
    _Float16* Wr3h = (_Float16*)(ws + o); o += w3B;
    (void)ws_size; (void)n_in; (void)out_size;

    const int strips  = (N + 15) / 16;
    const int gblocks = (strips + 7) / 8;
    const int eBlk    = (E + 255) / 256;
    const int nBlk    = (N + 255) / 256;
    const int aggBlk  = (N + 7) / 8;          // 8 waves/block, 1 node/wave
    const int wcBlk   = (KDIM * KDIM + 255) / 256;
    const int wc3Blk  = (OPAD * KDIM + 255) / 256;

    // ---- CSR build (once; reused by all three layers)
    hipMemsetAsync(degi, 0, (size_t)N * sizeof(int), stream);
    degree_int_kernel<<<eBlk, 256, 0, stream>>>(dstp, degi, E);
    rdeg_kernel<<<nBlk, 256, 0, stream>>>(degi, rdeg, N);
    scan1_kernel<<<B, 256, 0, stream>>>(degi, offs, bsums, N);
    scan2_kernel<<<1, 256, 0, stream>>>(bsums, B);
    scan3_kernel<<<B, 256, 0, stream>>>(offs, bsums, cursor, N);
    fill_adj_kernel<<<eBlk, 256, 0, stream>>>(srcp, dstp, cursor, adj, E);

    // ---- weights -> f16 (W3 zero-padded to 48 rows)
    convert_w_kernel<<<wcBlk, 256, 0, stream>>>(Wl1, Wl1h, KDIM, KDIM);
    convert_w_kernel<<<wcBlk, 256, 0, stream>>>(Wr1, Wr1h, KDIM, KDIM);
    convert_w_kernel<<<wcBlk, 256, 0, stream>>>(Wl2, Wl2h, KDIM, KDIM);
    convert_w_kernel<<<wcBlk, 256, 0, stream>>>(Wr2, Wr2h, KDIM, KDIM);
    convert_w_kernel<<<wc3Blk, 256, 0, stream>>>(Wl3, Wl3h, OUT, OPAD);
    convert_w_kernel<<<wc3Blk, 256, 0, stream>>>(Wr3, Wr3h, OUT, OPAD);

    // ---- layer 1: conv -> LN -> ReLU
    aggregate_csr_kernel<<<aggBlk, 256, 0, stream>>>(x, adj, offs, rdeg, agg, N, E);
    gemm_ln_kernel<8, true><<<gblocks, 256, 0, stream>>>(
        agg, x, Wl1h, Wr1h, bl1, g1, b1, h1b, N, KDIM, KDIM);

    // ---- layer 2
    aggregate_csr_kernel<<<aggBlk, 256, 0, stream>>>(h1b, adj, offs, rdeg, agg, N, E);
    gemm_ln_kernel<8, true><<<gblocks, 256, 0, stream>>>(
        agg, h1b, Wl2h, Wr2h, bl2, g2, b2, h2b, N, KDIM, KDIM);

    // ---- layer 3: conv -> log_softmax (into d_out)
    aggregate_csr_kernel<<<aggBlk, 256, 0, stream>>>(h2b, adj, offs, rdeg, agg, N, E);
    gemm_ln_kernel<3, false><<<gblocks, 256, 0, stream>>>(
        agg, h2b, Wl3h, Wr3h, bl3, nullptr, nullptr, (float*)d_out, N, OUT, OUT);

    logsoftmax_kernel<<<(N + 7) / 8, 256, 0, stream>>>((float*)d_out, N, OUT);
}
